// ElectronicSchnet_65163243815296
// MI455X (gfx1250) — matrix operand order repaired and checked by reading, not compiled, not run on hardware
//
#include <hip/hip_runtime.h>
#include <hip/hip_bf16.h>

typedef __attribute__((ext_vector_type(16))) _Float16 v16h;
typedef __attribute__((ext_vector_type(8)))  _Float16 v8h;
typedef __attribute__((ext_vector_type(8)))  float    v8f;
typedef __attribute__((ext_vector_type(4)))  float    v4f;

#define N_ELEC 32
#define N_ALL  36
#define N_PAIR 1120   // 32*35
#define BASIS  32
#define KDIM   128
#define EDIM   128
#define HID_K  64
#define HID_O  128
#define N_INT  3
#define BATCH  512
#define NPT    70     // 1120/16 pair tiles

// f16 weight staging layout in d_ws (element = _Float16)
#define W1OFF 0
#define W2OFF (W1OFF + N_INT*HID_K*BASIS)   // 6144
#define EWOFF (W2OFF + N_INT*KDIM*HID_K)    // 30720
#define O1OFF (EWOFF + N_INT*KDIM*EDIM)     // 79872
#define O2OFF (O1OFF + N_INT*HID_O*KDIM)    // 129024

#define LOG2F_C 0.6931471805599453f

__device__ __forceinline__ float sspf(float x) {
    // softplus(x) - log(2); arg of log is in (1,2] -> plain __logf is safe & fast
    return fmaxf(x, 0.0f) + __logf(1.0f + __expf(-fabsf(x))) - LOG2F_C;
}

__device__ __forceinline__ v8f wmma_f16(v16h a, v16h b, v8f c) {
    return __builtin_amdgcn_wmma_f32_16x16x32_f16(
        /*neg_a=*/false, a, /*neg_b=*/false, b,
        /*c_mod=*/(short)0, c, /*reuse_a=*/false, /*reuse_b=*/false);
}

// A-fragment (16x32 f16 tile, row-major source, row stride in elems).
// ISA 7.12.2: lane L -> M=L%16; halves 0..7 = K(8*hi+0..7), 8..15 = K(16+8*hi+0..7)
__device__ __forceinline__ v16h load_A16(const _Float16* base, int stride) {
    int lane = threadIdx.x & 31;
    int M = lane & 15, hi = lane >> 4;
    const _Float16* r = base + M * stride + 8 * hi;
    v8h p0 = *(const v8h*)(r);
    v8h p1 = *(const v8h*)(r + 16);
    v16h a;
#pragma unroll
    for (int c = 0; c < 8; ++c) { a[c] = p0[c]; a[8 + c] = p1[c]; }
    return a;
}

// B-fragment for GEMM  x . W^T : B[k][n] = W[n][k].
// base = &W[n0][k0], W row-major with row stride `stride`.
// lane L: n = n0 + (L&15); halves 0..7 -> K = k0 + 8*(L>>4) + idx, 8..15 -> K = k0 + 16 + 8*(L>>4) + idx-8
__device__ __forceinline__ v16h load_B16(const _Float16* base, int stride) {
    int lane = threadIdx.x & 31;
    int N = lane & 15, hs = lane >> 4;
    const _Float16* r = base + N * stride + 8 * hs;
    v8h p0 = *(const v8h*)(r);
    v8h p1 = *(const v8h*)(r + 16);
    v16h b;
#pragma unroll
    for (int c = 0; c < 8; ++c) { b[c] = p0[c]; b[8 + c] = p1[c]; }
    return b;
}

__global__ __launch_bounds__(256) void cvt_f32_f16(const float* __restrict__ src,
                                                   _Float16* __restrict__ dst, int n) {
    int i = blockIdx.x * 256 + threadIdx.x;
    if (i < n) dst[i] = (_Float16)src[i];
}

__global__ __launch_bounds__(256) void schnet_fused(
    const float* __restrict__ dists,      // (512,32,36,32)
    const float* __restrict__ emb_elec,   // (32,128)
    const float* __restrict__ emb_nuc,    // (4,128)
    const float* __restrict__ b1g,        // (3,64)
    const float* __restrict__ b2g,        // (3,128)
    const float* __restrict__ bo1g,       // (3,128)
    const float* __restrict__ bo2g,       // (3,128)
    const _Float16* __restrict__ wsH,     // f16 weight staging
    float* __restrict__ out)              // (512,32,128)
{
    __shared__ float    s_zs[N_ALL * KDIM];     // 18.4 KB  (elec rows 0..31, nuc 32..35)
    __shared__ float    s_z[N_ELEC * KDIM];     // 16 KB    reduction accumulator
    __shared__ float    s_xs[N_ELEC * EDIM];    // 16 KB    residual state (f32)
    __shared__ _Float16 s_xs16[N_ELEC * EDIM];  // 8 KB     f16 mirror for WMMA A
    __shared__ _Float16 s_stage[N_ELEC * KDIM]; // 8 KB     z16, then h2 staging
    __shared__ _Float16 s_H[8][16 * HID_K];     // 16 KB    per-wave H tile restripe

    const int tid  = threadIdx.x;
    const int wave = tid >> 5;
    const int lane = tid & 31;
    const int lo   = lane & 15;
    const int hi   = lane >> 4;
    const int b    = blockIdx.x;

    // ---- init: xs = broadcast(embedding_elec); zs nuc rows fixed across t
    for (int idx = tid; idx < N_ELEC * EDIM; idx += 256) {
        float v = emb_elec[idx];
        s_xs[idx] = v;
        s_xs16[idx] = (_Float16)v;
    }
    for (int idx = tid; idx < 4 * KDIM; idx += 256)
        s_zs[N_ELEC * KDIM + idx] = emb_nuc[idx];
    __syncthreads();

    for (int t = 0; t < N_INT; ++t) {
        const _Float16* w1t = wsH + W1OFF + t * (HID_K * BASIS);
        const _Float16* w2t = wsH + W2OFF + t * (KDIM * HID_K);
        const _Float16* ewt = wsH + EWOFF + t * (KDIM * EDIM);
        const _Float16* o1t = wsH + O1OFF + t * (HID_O * KDIM);
        const _Float16* o2t = wsH + O2OFF + t * (EDIM * HID_O);

        // ---- Phase A: zs[0:32] = xs @ ew^T ; zero z
        for (int idx = tid; idx < N_ELEC * KDIM; idx += 256) s_z[idx] = 0.0f;
#pragma unroll
        for (int q = 0; q < 2; ++q) {
            int tile = wave * 2 + q;
            int mt = tile >> 3, nt = tile & 7;
            v8f acc = {};
#pragma unroll
            for (int ks = 0; ks < 4; ++ks) {
                v16h a  = load_A16(&s_xs16[(mt * 16) * EDIM + ks * 32], EDIM);
                v16h bb = load_B16(&ewt[(nt * 16) * EDIM + ks * 32], EDIM);
                acc = wmma_f16(a, bb, acc);
            }
#pragma unroll
            for (int v = 0; v < 8; ++v)
                s_zs[(mt * 16 + v + 8 * hi) * KDIM + nt * 16 + lo] = acc[v];
        }
        __syncthreads();

        // ---- Phase B: stream 70 pair tiles; H -> Ws -> z accumulation
        float b1v[4], b2v[8];
#pragma unroll
        for (int n = 0; n < 4; ++n) b1v[n] = b1g[t * HID_K + n * 16 + lo];
#pragma unroll
        for (int n = 0; n < 8; ++n) b2v[n] = b2g[t * KDIM + n * 16 + lo];

        // w1 B-fragments are invariant across the 70 tiles: keep in registers
        v16h w1f[4];
#pragma unroll
        for (int n = 0; n < 4; ++n) w1f[n] = load_B16(&w1t[(n * 16) * BASIS], BASIS);

        for (int tt = wave; tt < NPT; tt += 8) {
            // gather db A-fragment straight from HBM (f32 -> f16)
            int pM = tt * 16 + lo;
            int iM = pM / 35;
            int jjM = pM - iM * 35;
            int jM = jjM + (jjM >= iM);
            const float* rb = dists + ((((size_t)b) * N_ELEC + iM) * N_ALL + jM) * BASIS + 8 * hi;
            v4f x0 = *(const v4f*)(rb);
            v4f x1 = *(const v4f*)(rb + 4);
            v4f x2 = *(const v4f*)(rb + 16);
            v4f x3 = *(const v4f*)(rb + 20);
            v16h adb;
#pragma unroll
            for (int c = 0; c < 4; ++c) {
                adb[c]      = (_Float16)x0[c];
                adb[4 + c]  = (_Float16)x1[c];
                adb[8 + c]  = (_Float16)x2[c];
                adb[12 + c] = (_Float16)x3[c];
            }
            // step1: H = ssp(db @ w1^T + b1)  (K=32, one WMMA per n-tile)
#pragma unroll
            for (int nt = 0; nt < 4; ++nt) {
                v8f acc = {};
                acc = wmma_f16(adb, w1f[nt], acc);
#pragma unroll
                for (int v = 0; v < 8; ++v) {
                    float hv = sspf(acc[v] + b1v[nt]);
                    s_H[wave][(v + 8 * hi) * HID_K + nt * 16 + lo] = (_Float16)hv;
                }
            }
            // same-wave LDS RAW: LDS is in-order per wave; fence the compiler + counter
            asm volatile("s_wait_dscnt 0" ::: "memory");
            v16h a0 = load_A16(&s_H[wave][0], HID_K);
            v16h a1 = load_A16(&s_H[wave][32], HID_K);

            // The 8 D-rows p = base..base+7 of this lane span at most TWO electrons i:
            // reduce in registers, then 1-2 LDS atomics per n-tile instead of 8.
            int baseP = tt * 16 + 8 * hi;
            int i0 = baseP / 35;
            int rem = baseP - i0 * 35;
            int vsplit = 35 - rem;   // rows v >= vsplit belong to electron i0+1

            // step2: Ws = H @ w2^T + b2 ; fused j-reduction into z
#pragma unroll
            for (int nt = 0; nt < 8; ++nt) {
                v8f acc = {};
                acc = wmma_f16(a0, load_B16(&w2t[(nt * 16) * HID_K], HID_K), acc);
                acc = wmma_f16(a1, load_B16(&w2t[(nt * 16) * HID_K + 32], HID_K), acc);
                int k = nt * 16 + lo;
                float acc0 = 0.0f, acc1 = 0.0f;
#pragma unroll
                for (int v = 0; v < 8; ++v) {
                    bool up = (v >= vsplit);
                    int iv = i0 + (up ? 1 : 0);
                    int jj = rem + v - (up ? 35 : 0);
                    int j = jj + (jj >= iv);
                    float term = (acc[v] + b2v[nt]) * s_zs[j * KDIM + k];
                    if (up) acc1 += term; else acc0 += term;
                }
                atomicAdd(&s_z[i0 * KDIM + k], acc0);
                if (vsplit < 8)
                    atomicAdd(&s_z[(i0 + 1) * KDIM + k], acc1);
            }
        }
        __syncthreads();

        // ---- Phase C: z -> f16 row-major staging (A operand of h2 GEMM)
        for (int idx = tid; idx < N_ELEC * KDIM; idx += 256)
            s_stage[idx] = (_Float16)s_z[idx];
        __syncthreads();

        // ---- Phase D: h2 = ssp(z @ o1^T + bo1)
        float h2r[2][8];
#pragma unroll
        for (int q = 0; q < 2; ++q) {
            int tile = wave * 2 + q;
            int mt = tile >> 3, nt = tile & 7;
            v8f acc = {};
#pragma unroll
            for (int ks = 0; ks < 4; ++ks) {
                v16h a  = load_A16(&s_stage[(mt * 16) * KDIM + ks * 32], KDIM);
                v16h bb = load_B16(&o1t[(nt * 16) * KDIM + ks * 32], KDIM);
                acc = wmma_f16(a, bb, acc);
            }
            float bo1v = bo1g[t * HID_O + nt * 16 + lo];
#pragma unroll
            for (int v = 0; v < 8; ++v) h2r[q][v] = sspf(acc[v] + bo1v);
        }
        __syncthreads();   // all z16 reads done before overwriting s_stage
#pragma unroll
        for (int q = 0; q < 2; ++q) {
            int tile = wave * 2 + q;
            int mt = tile >> 3, nt = tile & 7;
#pragma unroll
            for (int v = 0; v < 8; ++v)
                s_stage[(mt * 16 + v + 8 * hi) * HID_O + nt * 16 + lo] = (_Float16)h2r[q][v];
        }
        __syncthreads();

        // ---- Phase E: xs += h2 @ o2^T + bo2
#pragma unroll
        for (int q = 0; q < 2; ++q) {
            int tile = wave * 2 + q;
            int mt = tile >> 3, nt = tile & 7;
            v8f acc = {};
#pragma unroll
            for (int ks = 0; ks < 4; ++ks) {
                v16h a  = load_A16(&s_stage[(mt * 16) * HID_O + ks * 32], HID_O);
                v16h bb = load_B16(&o2t[(nt * 16) * HID_O + ks * 32], HID_O);
                acc = wmma_f16(a, bb, acc);
            }
            float bo2v = bo2g[t * EDIM + nt * 16 + lo];
#pragma unroll
            for (int v = 0; v < 8; ++v) {
                int r = mt * 16 + v + 8 * hi;
                int c = nt * 16 + lo;
                float nv = s_xs[r * EDIM + c] + acc[v] + bo2v;
                s_xs[r * EDIM + c] = nv;
                s_xs16[r * EDIM + c] = (_Float16)nv;
            }
        }
        __syncthreads();
    }

    for (int idx = tid; idx < N_ELEC * EDIM; idx += 256)
        out[(size_t)b * (N_ELEC * EDIM) + idx] = s_xs[idx];
}

extern "C" void kernel_launch(void* const* d_in, const int* in_sizes, int n_in,
                              void* d_out, int out_size, void* d_ws, size_t ws_size,
                              hipStream_t stream) {
    const float* dists    = (const float*)d_in[0];   // dists_basis
    const float* emb_elec = (const float*)d_in[1];
    const float* emb_nuc  = (const float*)d_in[2];
    const float* w1       = (const float*)d_in[3];   // kernel_w1
    const float* b1       = (const float*)d_in[4];
    const float* w2       = (const float*)d_in[5];
    const float* b2       = (const float*)d_in[6];
    const float* ew       = (const float*)d_in[7];   // embed_in_w
    const float* o1       = (const float*)d_in[8];   // embed_out_w1
    const float* bo1      = (const float*)d_in[9];
    const float* o2       = (const float*)d_in[10];  // embed_out_w2
    const float* bo2      = (const float*)d_in[11];
    float* out = (float*)d_out;

    _Float16* wsH = (_Float16*)d_ws;

    // one-time (per launch) f32 -> f16 weight staging; L2-resident afterwards
    auto cvt = [&](const float* s, _Float16* d, int n) {
        cvt_f32_f16<<<(n + 255) / 256, 256, 0, stream>>>(s, d, n);
    };
    cvt(w1, wsH + W1OFF, N_INT * HID_K * BASIS);
    cvt(w2, wsH + W2OFF, N_INT * KDIM * HID_K);
    cvt(ew, wsH + EWOFF, N_INT * KDIM * EDIM);
    cvt(o1, wsH + O1OFF, N_INT * HID_O * KDIM);
    cvt(o2, wsH + O2OFF, N_INT * EDIM * HID_O);

    schnet_fused<<<BATCH, 256, 0, stream>>>(dists, emb_elec, emb_nuc,
                                            b1, b2, bo1, bo2, wsH, out);
}
